// MOEBlock_146028888420
// MI455X (gfx1250) — compile-verified, matrix operand order checked by reading, not run on hardware
//
#include <hip/hip_runtime.h>
#include <math.h>

// ---------------- problem constants ----------------
#define T_TOK 8192      // 4*2048 tokens
#define H_DIM 512       // hidden
#define I_DIM 384       // moe intermediate
#define N1_DIM 768      // 2*I
#define E_NUM 16        // routed experts
#define ALPHA_C 1.702f
#define LIMIT_C 7.0f

typedef __bf16 bf16;
typedef __bf16 v16bf __attribute__((ext_vector_type(16)));
typedef __bf16 v8bf  __attribute__((ext_vector_type(8)));
typedef float  v8f   __attribute__((ext_vector_type(8)));

// Build a v16bf A/B fragment for wmma_f32_16x16x32_bf16 from an LDS row of 32
// bf16 K-values, following the ISA 16-bit 16x32 layout:
//   lanes 0-15  hold K = {0..7, 16..23}
//   lanes 16-31 hold K = {8..15, 24..31}
__device__ __forceinline__ v16bf frag_ld(const bf16* rowBase, int half) {
  v8bf lo = *(const v8bf*)(rowBase + half * 8);
  v8bf hi = *(const v8bf*)(rowBase + 16 + half * 8);
  return __builtin_shufflevector(lo, hi, 0,1,2,3,4,5,6,7,8,9,10,11,12,13,14,15);
}

// ---------------- f32 -> bf16 convert ----------------
__global__ __launch_bounds__(256) void cvt_kernel(const float* __restrict__ s,
                                                  bf16* __restrict__ d, int n) {
  int i = blockIdx.x * 256 + threadIdx.x;
  if (i < n) d[i] = (bf16)s[i];
}

__global__ void moe_init(int* counts) {
  if (threadIdx.x < E_NUM) counts[threadIdx.x] = 0;
}

// ---------------- gating: logits, softmax, top-2, dispatch lists ------------
__global__ __launch_bounds__(256) void moe_gate(
    const float* __restrict__ x, const float* __restrict__ gw,
    const float* __restrict__ gb, bf16* __restrict__ xb,
    int* __restrict__ counts, int* __restrict__ rows,
    float* __restrict__ topw, float* __restrict__ partP,
    float* __restrict__ partImp) {
  __shared__ float gwS[E_NUM * H_DIM];        // 32 KB
  __shared__ float red[E_NUM][264];           // block reduction scratch
  const int tid = threadIdx.x;
  const int t = blockIdx.x * 256 + tid;

  for (int i = tid; i < E_NUM * H_DIM; i += 256) gwS[i] = gw[i];
  __syncthreads();

  float lg[E_NUM];
#pragma unroll
  for (int e = 0; e < E_NUM; ++e) lg[e] = gb[e];

  const float* xr = x + (size_t)t * H_DIM;
  bf16* xbr = xb + (size_t)t * H_DIM;
  for (int k = 0; k < H_DIM; k += 4) {
    float4 xv = *(const float4*)(xr + k);
#pragma unroll
    for (int e = 0; e < E_NUM; ++e) {
      const float* g = gwS + e * H_DIM + k;
      lg[e] += xv.x * g[0] + xv.y * g[1] + xv.z * g[2] + xv.w * g[3];
    }
    xbr[k]     = (bf16)xv.x;
    xbr[k + 1] = (bf16)xv.y;
    xbr[k + 2] = (bf16)xv.z;
    xbr[k + 3] = (bf16)xv.w;
  }

  // softmax
  float mx = lg[0];
#pragma unroll
  for (int e = 1; e < E_NUM; ++e) mx = fmaxf(mx, lg[e]);
  float pr[E_NUM];
  float sum = 0.f;
#pragma unroll
  for (int e = 0; e < E_NUM; ++e) { pr[e] = expf(lg[e] - mx); sum += pr[e]; }
  float inv = 1.f / sum;
#pragma unroll
  for (int e = 0; e < E_NUM; ++e) pr[e] *= inv;

  // top-2 (ties -> lowest index, matches lax.top_k)
  int i1 = 0; float p1 = pr[0];
#pragma unroll
  for (int e = 1; e < E_NUM; ++e) if (pr[e] > p1) { p1 = pr[e]; i1 = e; }
  int i2 = (i1 == 0) ? 1 : 0; float p2 = pr[i2];
#pragma unroll
  for (int e = 0; e < E_NUM; ++e)
    if (e != i1 && e != i2 && pr[e] > p2) { p2 = pr[e]; i2 = e; }

  float s12 = p1 + p2;
  int s1 = atomicAdd(&counts[i1], 1);
  rows[i1 * T_TOK + s1] = t * 2 + 0;
  topw[t * 2 + 0] = p1 / s12;
  int s2 = atomicAdd(&counts[i2], 1);
  rows[i2 * T_TOK + s2] = t * 2 + 1;
  topw[t * 2 + 1] = p2 / s12;

  // deterministic block-level reduction of probs (for P) and logits (importance)
#pragma unroll
  for (int e = 0; e < E_NUM; ++e) red[e][tid] = pr[e];
  for (int s = 128; s > 0; s >>= 1) {
    __syncthreads();
    if (tid < s) {
#pragma unroll
      for (int e = 0; e < E_NUM; ++e) red[e][tid] += red[e][tid + s];
    }
  }
  __syncthreads();
  if (tid < E_NUM) partP[blockIdx.x * E_NUM + tid] = red[tid][0];
  __syncthreads();
#pragma unroll
  for (int e = 0; e < E_NUM; ++e) red[e][tid] = lg[e];
  for (int s = 128; s > 0; s >>= 1) {
    __syncthreads();
    if (tid < s) {
#pragma unroll
      for (int e = 0; e < E_NUM; ++e) red[e][tid] += red[e][tid + s];
    }
  }
  __syncthreads();
  if (tid < E_NUM) partImp[blockIdx.x * E_NUM + tid] = red[tid][0];
}

// ---------------- aux loss ----------------
__global__ void moe_aux(const float* __restrict__ partP,
                        const float* __restrict__ partImp,
                        const int* __restrict__ counts,
                        float* __restrict__ outAux) {
  __shared__ float P[E_NUM], IMP[E_NUM];
  int e = threadIdx.x;
  if (e < E_NUM) {
    float sp = 0.f, si = 0.f;
    for (int b = 0; b < T_TOK / 256; ++b) {
      sp += partP[b * E_NUM + e];
      si += partImp[b * E_NUM + e];
    }
    P[e] = sp / (float)T_TOK;
    IMP[e] = si;
  }
  __syncthreads();
  if (e == 0) {
    float lb = 0.f, mean = 0.f;
    for (int i = 0; i < E_NUM; ++i) {
      lb += P[i] * ((float)counts[i] / (float)(T_TOK * 2));
      mean += IMP[i];
    }
    mean /= (float)E_NUM;
    float var = 0.f;
    for (int i = 0; i < E_NUM; ++i) {
      float d = IMP[i] - mean;
      var += d * d;
    }
    var /= (float)(E_NUM - 1);                 // ddof=1
    float cv = sqrtf(var) / (mean + 1e-6f);
    float loss_lb = 0.01f * (float)E_NUM * lb; // W_LOAD * E * sum(P*D)
    float loss_imp = 0.01f * cv * cv;          // W_IMP * cv^2
    outAux[0] = 0.01f * (loss_lb + loss_imp);  // W_AUX * (...)
  }
}

// ---------------- GEMM1: [rows x 512] @ W1^T[512 x 768] -> swiglu -> act ----
// block: 64 rows x 128 n1-cols, 256 threads (8 waves: 2M x 4N of 32x32)
__global__ __launch_bounds__(256) void moe_gemm1(
    const bf16* __restrict__ xb, const bf16* __restrict__ w1,
    const float* __restrict__ b1, bf16* __restrict__ act,
    const int* __restrict__ rowsList, const int* __restrict__ cnt) {
  __shared__ bf16 At[64][32];
  __shared__ bf16 Bt[128][32];
  __shared__ float Yt[64][129];

  const int count = rowsList ? cnt[0] : T_TOK;
  const int m0 = blockIdx.x * 64;
  if (m0 >= count) return;
  const int n0 = blockIdx.y * 128;
  const int tid = threadIdx.x;
  const int lane = tid & 31, wid = tid >> 5;
  const int half = lane >> 4, lr = lane & 15;
  const int wm = wid & 1, wn = wid >> 1;

  v8f acc[2][2] = {};

  const int ar = tid >> 2;        // 0..63
  const int ak = (tid & 3) * 8;   // 0,8,16,24
  const int rgA = m0 + ar;
  const int tokA = (rgA < count) ? (rowsList ? (rowsList[rgA] >> 1) : rgA)
                                 : (rowsList ? (rowsList[m0] >> 1) : m0);
  const int br = tid >> 1;        // 0..127
  const int bk = (tid & 1) * 16;  // 0 or 16

  for (int k0 = 0; k0 < H_DIM; k0 += 32) {
    *(uint4*)(&At[ar][ak]) = *(const uint4*)(xb + (size_t)tokA * H_DIM + k0 + ak);
    const bf16* wrow = w1 + (size_t)(n0 + br) * H_DIM + k0 + bk;
    *(uint4*)(&Bt[br][bk]) = *(const uint4*)(wrow);
    *(uint4*)(&Bt[br][bk + 8]) = *(const uint4*)(wrow + 8);
    __syncthreads();
    v16bf af[2], bfr[2];
#pragma unroll
    for (int m = 0; m < 2; ++m) af[m] = frag_ld(&At[wm * 32 + m * 16 + lr][0], half);
#pragma unroll
    for (int n = 0; n < 2; ++n) bfr[n] = frag_ld(&Bt[wn * 32 + n * 16 + lr][0], half);
#pragma unroll
    for (int m = 0; m < 2; ++m)
#pragma unroll
      for (int n = 0; n < 2; ++n)
        acc[m][n] = __builtin_amdgcn_wmma_f32_16x16x32_bf16(
            false, af[m], false, bfr[n], (short)0, acc[m][n], false, false);
    __syncthreads();
  }

  // C layout: VGPR v -> M = v (lanes 0-15) / v+8 (lanes 16-31); N = lane&15
#pragma unroll
  for (int m = 0; m < 2; ++m)
#pragma unroll
    for (int n = 0; n < 2; ++n)
#pragma unroll
      for (int v = 0; v < 8; ++v) {
        int r = wm * 32 + m * 16 + half * 8 + v;
        int c = wn * 32 + n * 16 + lr;
        Yt[r][c] = acc[m][n][v] + b1[n0 + c];
      }
  __syncthreads();

  // interleaved swiglu: even col -> gate, odd col -> linear
  for (int i = tid; i < 64 * 64; i += 256) {
    int r = i >> 6, j = i & 63;
    if (m0 + r < count) {
      float g = Yt[r][2 * j];
      float l = Yt[r][2 * j + 1];
      g = fminf(fmaxf(g, -LIMIT_C), LIMIT_C);
      l = fminf(fmaxf(l, -LIMIT_C), LIMIT_C);
      float sg = 1.0f / (1.0f + expf(-ALPHA_C * g));
      float val = g * sg + (l + 1.0f);
      act[(size_t)(m0 + r) * I_DIM + (n0 >> 1) + j] = (bf16)val;
    }
  }
}

// ---------------- GEMM2: [rows x 384] @ W2^T[384 x 512] -> scatter to out ---
__global__ __launch_bounds__(256) void moe_gemm2(
    const bf16* __restrict__ act, const bf16* __restrict__ w2,
    const float* __restrict__ b2, const float* __restrict__ topw,
    float* __restrict__ out, const int* __restrict__ rowsList,
    const int* __restrict__ cnt) {
  __shared__ bf16 At[64][32];
  __shared__ bf16 Bt[128][32];

  const int count = rowsList ? cnt[0] : T_TOK;
  const int m0 = blockIdx.x * 64;
  if (m0 >= count) return;
  const int n0 = blockIdx.y * 128;
  const int tid = threadIdx.x;
  const int lane = tid & 31, wid = tid >> 5;
  const int half = lane >> 4, lr = lane & 15;
  const int wm = wid & 1, wn = wid >> 1;

  v8f acc[2][2] = {};

  const int ar = tid >> 2;
  const int ak = (tid & 3) * 8;
  const int rsrc = (m0 + ar < count) ? (m0 + ar) : m0;
  const int br = tid >> 1;
  const int bk = (tid & 1) * 16;

  for (int k0 = 0; k0 < I_DIM; k0 += 32) {
    *(uint4*)(&At[ar][ak]) = *(const uint4*)(act + (size_t)rsrc * I_DIM + k0 + ak);
    const bf16* wrow = w2 + (size_t)(n0 + br) * I_DIM + k0 + bk;
    *(uint4*)(&Bt[br][bk]) = *(const uint4*)(wrow);
    *(uint4*)(&Bt[br][bk + 8]) = *(const uint4*)(wrow + 8);
    __syncthreads();
    v16bf af[2], bfr[2];
#pragma unroll
    for (int m = 0; m < 2; ++m) af[m] = frag_ld(&At[wm * 32 + m * 16 + lr][0], half);
#pragma unroll
    for (int n = 0; n < 2; ++n) bfr[n] = frag_ld(&Bt[wn * 32 + n * 16 + lr][0], half);
#pragma unroll
    for (int m = 0; m < 2; ++m)
#pragma unroll
      for (int n = 0; n < 2; ++n)
        acc[m][n] = __builtin_amdgcn_wmma_f32_16x16x32_bf16(
            false, af[m], false, bfr[n], (short)0, acc[m][n], false, false);
    __syncthreads();
  }

#pragma unroll
  for (int m = 0; m < 2; ++m)
#pragma unroll
    for (int n = 0; n < 2; ++n)
#pragma unroll
      for (int v = 0; v < 8; ++v) {
        int rg = m0 + wm * 32 + m * 16 + half * 8 + v;
        if (rg < count) {
          int c = n0 + wn * 32 + n * 16 + lr;
          float val = acc[m][n][v] + b2[c];
          if (rowsList) {
            int entry = rowsList[rg];
            int tok = entry >> 1;
            out[(size_t)tok * H_DIM + c] += topw[entry] * val;  // expert accumulate
          } else {
            out[(size_t)rg * H_DIM + c] = val;                  // shared: init
          }
        }
      }
}

// ---------------- host-side orchestration ----------------
extern "C" void kernel_launch(void* const* d_in, const int* in_sizes, int n_in,
                              void* d_out, int out_size, void* d_ws, size_t ws_size,
                              hipStream_t stream) {
  (void)in_sizes; (void)n_in; (void)out_size; (void)ws_size;
  const float* x     = (const float*)d_in[0];
  const float* gw    = (const float*)d_in[1];
  const float* gb    = (const float*)d_in[2];
  const float* sh_w1 = (const float*)d_in[3];
  const float* sh_b1 = (const float*)d_in[4];
  const float* sh_w2 = (const float*)d_in[5];
  const float* sh_b2 = (const float*)d_in[6];
  const float* ex_w1 = (const float*)d_in[7];
  const float* ex_b1 = (const float*)d_in[8];
  const float* ex_w2 = (const float*)d_in[9];
  const float* ex_b2 = (const float*)d_in[10];
  float* out = (float*)d_out;

  // workspace carve-up (~35.5 MB total)
  char* ws = (char*)d_ws;
  size_t off = 0;
  auto alloc = [&](size_t bytes) -> char* {
    char* p = ws + off;
    off = (off + bytes + 255) & ~(size_t)255;
    return p;
  };
  bf16* xb      = (bf16*)alloc((size_t)T_TOK * H_DIM * 2);
  bf16* w1b     = (bf16*)alloc((size_t)(E_NUM + 1) * N1_DIM * H_DIM * 2);
  bf16* w2b     = (bf16*)alloc((size_t)(E_NUM + 1) * H_DIM * I_DIM * 2);
  bf16* actBuf  = (bf16*)alloc((size_t)T_TOK * I_DIM * 2);
  int*  rowsL   = (int*)alloc((size_t)E_NUM * T_TOK * 4);
  float* topw   = (float*)alloc((size_t)T_TOK * 2 * 4);
  int*  counts  = (int*)alloc(64);
  float* partP  = (float*)alloc((T_TOK / 256) * E_NUM * 4);
  float* partImp= (float*)alloc((T_TOK / 256) * E_NUM * 4);

  // weights -> bf16 (shared at slot 0, expert e at slot 1+e)
  {
    int n;
    n = N1_DIM * H_DIM;
    cvt_kernel<<<(n + 255) / 256, 256, 0, stream>>>(sh_w1, w1b, n);
    n = E_NUM * N1_DIM * H_DIM;
    cvt_kernel<<<(n + 255) / 256, 256, 0, stream>>>(ex_w1, w1b + N1_DIM * H_DIM, n);
    n = H_DIM * I_DIM;
    cvt_kernel<<<(n + 255) / 256, 256, 0, stream>>>(sh_w2, w2b, n);
    n = E_NUM * H_DIM * I_DIM;
    cvt_kernel<<<(n + 255) / 256, 256, 0, stream>>>(ex_w2, w2b + H_DIM * I_DIM, n);
  }

  moe_init<<<1, 32, 0, stream>>>(counts);
  moe_gate<<<T_TOK / 256, 256, 0, stream>>>(x, gw, gb, xb, counts, rowsL, topw,
                                            partP, partImp);
  moe_aux<<<1, 32, 0, stream>>>(partP, partImp, counts, out + (size_t)T_TOK * H_DIM);

  dim3 g1(T_TOK / 64, N1_DIM / 128);  // 128 x 6
  dim3 g2(T_TOK / 64, H_DIM / 128);   // 128 x 4

  // shared expert first: initializes out with '='
  moe_gemm1<<<g1, 256, 0, stream>>>(xb, w1b, sh_b1, actBuf, nullptr, nullptr);
  moe_gemm2<<<g2, 256, 0, stream>>>(actBuf, w2b, sh_b2, topw, out, nullptr, nullptr);

  // routed experts: gather rows, accumulate w * expert(x) into out
  for (int e = 0; e < E_NUM; ++e) {
    moe_gemm1<<<g1, 256, 0, stream>>>(
        xb, w1b + (size_t)(1 + e) * N1_DIM * H_DIM, ex_b1 + (size_t)e * N1_DIM,
        actBuf, rowsL + (size_t)e * T_TOK, counts + e);
    moe_gemm2<<<g2, 256, 0, stream>>>(
        actBuf, w2b + (size_t)(1 + e) * H_DIM * I_DIM, ex_b2 + (size_t)e * H_DIM,
        topw, out, rowsL + (size_t)e * T_TOK, counts + e);
  }
}
